// LinearAttention_40175124087245
// MI455X (gfx1250) — matrix operand order repaired and checked by reading, not compiled, not run on hardware
//
#include <hip/hip_runtime.h>

// ---------------- problem constants ----------------
#define B_   4
#define S_   8192
#define D_   512
#define H_   8
#define HD_  64
#define TD_  1536          // 3*D
#define BS_  (B_*S_)       // 32768
#define LDP_ 40            // LDS row pitch in f16 (80B = 5x16B chunks, 20-bank stride)

typedef __attribute__((ext_vector_type(16))) _Float16 v16h;
typedef __attribute__((ext_vector_type(8)))  float    v8f;

// elu(x)+1 feature map
__device__ __forceinline__ float phi_f(float x) {
    return x > 0.0f ? x + 1.0f : __expf(x);
}

__device__ __forceinline__ v8f wmma32(v16h a, v16h b, v8f c) {
    return __builtin_amdgcn_wmma_f32_16x16x32_f16(false, a, false, b, (short)0, c,
                                                  false, false);
}

// ---- CDNA5 async global->LDS copy (ASYNCcnt path, §15.18 / 08_async_tensor) ----
// Per-lane 16B transfer: VDST = LDS byte address (vgpr), VADDR = 64-bit global addr.
__device__ __forceinline__ void async_copy16(unsigned int lds_off, const void* g) {
    asm volatile("global_load_async_to_lds_b128 %0, %1, off"
                 :: "v"(lds_off), "v"((unsigned long long)(size_t)g)
                 : "memory");
}
__device__ __forceinline__ void wait_async0() {
    asm volatile("s_wait_asynccnt 0" ::: "memory");
}

// ---- WMMA fragment loaders (wave32, 16x16x32 f16) ----
// A (16x32): lane holds row M=lr; half j -> K = (j&7) + (j>=8?16:0) + hi*8
__device__ __forceinline__ v16h a_frag_f16(const _Float16* __restrict__ p0, size_t ld, int hi, int lr) {
    const _Float16* p = p0 + (size_t)lr * ld + hi * 8;
    v16h a;
#pragma unroll
    for (int j = 0; j < 8; ++j) a[j]     = p[j];
#pragma unroll
    for (int j = 0; j < 8; ++j) a[j + 8] = p[j + 16];
    return a;
}
// B (32x16): lane holds col N=lr; half j -> K = (j&7) + (j>=8?16:0) + hi*8.
// layout requirement: element (k,n) at p0 + n*ld + k  (k contiguous)
__device__ __forceinline__ v16h b_frag_f16(const _Float16* __restrict__ p0, size_t ld, int hi, int lr) {
    const _Float16* p = p0 + (size_t)lr * ld + hi * 8;
    v16h b;
#pragma unroll
    for (int j = 0; j < 8; ++j) b[j]     = p[j];
#pragma unroll
    for (int j = 0; j < 8; ++j) b[j + 8] = p[j + 16];
    return b;
}

// ---------------- f32 -> f16 conversion pass ----------------
__global__ __launch_bounds__(256) void k_cvt(const float* __restrict__ in,
                                             _Float16* __restrict__ out, int n) {
    int i = (blockIdx.x * 256 + threadIdx.x) * 4;
    if (i + 3 < n) {
        float4 v = *(const float4*)(in + i);
        out[i + 0] = (_Float16)v.x;
        out[i + 1] = (_Float16)v.y;
        out[i + 2] = (_Float16)v.z;
        out[i + 3] = (_Float16)v.w;
    }
}

// ---------------- shared 128x64 GEMM block body (f16 WMMA, async-LDS staged) ---------
// Block = 256 threads (8 waves, 4x2 wave grid), each wave: 32x32 via 2x2 accumulators.
// A: row-major (ldA), rows row0..row0+127.  B: element (k,n) at Bg + (col0+n)*ldB + k.
__device__ __forceinline__ void gemm_block_128x64(
    const _Float16* __restrict__ Ag, int ldA, int row0,
    const _Float16* __restrict__ Bg, int ldB, int col0,
    int kTot, v8f c[2][2])
{
    __shared__ _Float16 As[128][LDP_];   // 10.0 KB
    __shared__ _Float16 Bs[64][LDP_];    //  5.0 KB
    const int tid  = threadIdx.x;
    const int lane = tid & 31;
    const int warp = tid >> 5;
    const int hi = lane >> 4, lr = lane & 15;
    const int wm = warp >> 1;            // 0..3 -> M offset wm*32
    const int wn = warp & 1;             // 0..1 -> N offset wn*32

    const unsigned int asb = (unsigned int)(size_t)&As[0][0];
    const unsigned int bsb = (unsigned int)(size_t)&Bs[0][0];

    for (int kk = 0; kk < kTot; kk += 32) {
        // ---- stage A panel: 128 rows x 32 k = 512 x 16B chunks, 2 per thread ----
#pragma unroll
        for (int i = 0; i < 2; ++i) {
            const int ch = tid + i * 256;
            const int r = ch >> 2, c4 = ch & 3;
            async_copy16(asb + (unsigned int)(r * LDP_ + c4 * 8) * 2,
                         Ag + (size_t)(row0 + r) * ldA + kk + c4 * 8);
        }
        // ---- stage B panel: 64 rows x 32 k = 256 x 16B chunks, 1 per thread ----
        {
            const int r = tid >> 2, c4 = tid & 3;
            async_copy16(bsb + (unsigned int)(r * LDP_ + c4 * 8) * 2,
                         Bg + (size_t)(col0 + r) * ldB + kk + c4 * 8);
        }
        wait_async0();
        __syncthreads();

        v16h a0 = a_frag_f16(&As[wm * 32 +  0][0], LDP_, hi, lr);
        v16h a1 = a_frag_f16(&As[wm * 32 + 16][0], LDP_, hi, lr);
        v16h b0 = b_frag_f16(&Bs[wn * 32 +  0][0], LDP_, hi, lr);
        v16h b1 = b_frag_f16(&Bs[wn * 32 + 16][0], LDP_, hi, lr);
        c[0][0] = wmma32(a0, b0, c[0][0]);
        c[0][1] = wmma32(a0, b1, c[0][1]);
        c[1][0] = wmma32(a1, b0, c[1][0]);
        c[1][1] = wmma32(a1, b1, c[1][1]);
        __syncthreads();
    }
}

// ---------------- kernel 1: in_proj GEMM + bias + phi + mask -> q/kT/vT (f16) --------
__global__ __launch_bounds__(256) void k_qkv(
    const _Float16* __restrict__ Xh,        // (BS, D) f16
    const _Float16* __restrict__ Wih,       // (3D, D) f16
    const float* __restrict__ bi,           // (3D)
    const unsigned char* __restrict__ mask, // (B,S) nonzero = ignore
    _Float16* __restrict__ qh,              // [B][H][S][HD]   phi(q)
    _Float16* __restrict__ kT,              // [B][H][HD][S]   phi(k)*m
    _Float16* __restrict__ vT)              // [B][H][HD][S]   v*m
{
    const int NB = TD_ / 64;                // 24 col blocks
    const int nb = blockIdx.x % NB;
    const int mb = blockIdx.x / NB;
    const int row0 = mb * 128;
    const int col0 = nb * 64;

    v8f c[2][2] = {};
    gemm_block_128x64(Xh, D_, row0, Wih, D_, col0, D_, c);

    const int lane = threadIdx.x & 31;
    const int warp = threadIdx.x >> 5;
    const int hi = lane >> 4, lr = lane & 15;
    const int wm = warp >> 1, wn = warp & 1;

#pragma unroll
    for (int ni = 0; ni < 2; ++ni) {
        const int col = col0 + wn * 32 + ni * 16 + lr;
        const float bias = bi[col];
        const int sec = col / D_;           // 0=q 1=k 2=v
        const int cs  = col % D_;
        const int h   = cs / HD_;
        const int hd  = cs % HD_;
#pragma unroll
        for (int mi = 0; mi < 2; ++mi) {
#pragma unroll
            for (int r = 0; r < 8; ++r) {
                const int row = row0 + wm * 32 + mi * 16 + r + hi * 8;
                const int b_  = row / S_;
                const int s   = row % S_;
                float v = c[mi][ni][r] + bias;
                if (sec == 0) {
                    v = phi_f(v);
                    qh[(((size_t)b_ * H_ + h) * S_ + s) * HD_ + hd] = (_Float16)v;
                } else {
                    const float mv = mask[(size_t)b_ * S_ + s] ? 0.0f : 1.0f;
                    if (sec == 1) {
                        v = phi_f(v) * mv;
                        kT[(((size_t)b_ * H_ + h) * HD_ + hd) * S_ + s] = (_Float16)v;
                    } else {
                        v = v * mv;
                        vT[(((size_t)b_ * H_ + h) * HD_ + hd) * S_ + s] = (_Float16)v;
                    }
                }
            }
        }
    }
}

// ---------------- kernel 2: ksum[b,h,d] = sum_s phi(k) ----------------
__global__ __launch_bounds__(256) void k_ksum(const _Float16* __restrict__ kT,
                                              float* __restrict__ ksum) {
    __shared__ float red[256];
    const int bh = blockIdx.x;               // 0..31
    const int t  = threadIdx.x;
    const int d    = t & 63;
    const int part = t >> 6;                 // 4 chunks over S
    const _Float16* p = kT + ((size_t)bh * HD_ + d) * S_ + (size_t)part * (S_ / 4);
    float acc = 0.f;
    for (int s = 0; s < S_ / 4; ++s) acc += (float)p[s];
    red[t] = acc;
    __syncthreads();
    if (part == 0)
        ksum[(size_t)bh * HD_ + d] = red[d] + red[d + 64] + red[d + 128] + red[d + 192];
}

// ---------------- kernel 3: kv[b,h,d,e] = sum_s kT[d,s]*vT[e,s] (stored [e][d]) ------
__global__ __launch_bounds__(512) void k_kv(const _Float16* __restrict__ kT,
                                            const _Float16* __restrict__ vT,
                                            _Float16* __restrict__ kvT) {
    const int bh = blockIdx.x;
    const int lane = threadIdx.x & 31;
    const int warp = threadIdx.x >> 5;       // 16 waves: one 16x16 tile each
    const int hi = lane >> 4, lr = lane & 15;
    const int d0 = (warp >> 2) * 16;
    const int e0 = (warp & 3)  * 16;
    const _Float16* kp = kT + ((size_t)bh * HD_ + d0) * S_;   // A: rows=d, K=s
    const _Float16* vp = vT + ((size_t)bh * HD_ + e0) * S_;   // B: cols=e, K=s
    v8f c = {};
    for (int s0 = 0; s0 < S_; s0 += 32) {
        v16h a = a_frag_f16(kp + s0, S_, hi, lr);
        v16h b = b_frag_f16(vp + s0, S_, hi, lr);
        c = wmma32(a, b, c);
    }
#pragma unroll
    for (int r = 0; r < 8; ++r) {
        const int d = d0 + r + hi * 8;
        const int e = e0 + lr;
        kvT[((size_t)bh * HD_ + e) * HD_ + d] = (_Float16)c[r];
    }
}

// ---------------- kernel 4: denom[b,h,s] = max(q . ksum, 1e-6) ----------------
__global__ __launch_bounds__(256) void k_denom(const _Float16* __restrict__ qh,
                                               const float* __restrict__ ksum,
                                               float* __restrict__ denom) {
    const size_t i  = (size_t)blockIdx.x * blockDim.x + threadIdx.x;  // < B*H*S
    const size_t bh = i / S_;
    const _Float16* q = qh + i * HD_;
    const float* ks = ksum + bh * HD_;
    float acc = 0.f;
#pragma unroll
    for (int d = 0; d < HD_; ++d) acc += (float)q[d] * ks[d];
    denom[i] = fmaxf(acc, 1e-6f);
}

// ---------------- kernel 5: numerator GEMM + divide -> ctx (B,S,D) f16 ----------------
__global__ __launch_bounds__(256) void k_attn(const _Float16* __restrict__ qh,
                                              const _Float16* __restrict__ kvT,
                                              const float* __restrict__ denom,
                                              _Float16* __restrict__ ctx) {
    const int lane = threadIdx.x & 31;
    const int warp = threadIdx.x >> 5;
    const int hi = lane >> 4, lr = lane & 15;
    const int gw = blockIdx.x * 8 + warp;    // < B*H*(S/16)
    const int bh = gw / (S_ / 16);
    const int mt = gw % (S_ / 16);
    const int s0 = mt * 16;
    const int b_ = bh / H_;
    const int h  = bh % H_;

    const _Float16* qp  = qh  + ((size_t)bh * S_ + s0) * HD_;
    const _Float16* kvp = kvT + (size_t)bh * HD_ * HD_;       // [e][d], d contiguous

    v8f c0 = {}, c1 = {}, c2 = {}, c3 = {};
#pragma unroll
    for (int kk = 0; kk < HD_; kk += 32) {
        v16h a  = a_frag_f16(qp + kk, HD_, hi, lr);
        v16h b0 = b_frag_f16(kvp + (size_t)(0 * 16) * HD_ + kk, HD_, hi, lr);
        v16h b1 = b_frag_f16(kvp + (size_t)(1 * 16) * HD_ + kk, HD_, hi, lr);
        v16h b2 = b_frag_f16(kvp + (size_t)(2 * 16) * HD_ + kk, HD_, hi, lr);
        v16h b3 = b_frag_f16(kvp + (size_t)(3 * 16) * HD_ + kk, HD_, hi, lr);
        c0 = wmma32(a, b0, c0);
        c1 = wmma32(a, b1, c1);
        c2 = wmma32(a, b2, c2);
        c3 = wmma32(a, b3, c3);
    }
#pragma unroll
    for (int r = 0; r < 8; ++r) {
        const int s = s0 + r + hi * 8;
        const float inv = 1.0f / denom[(size_t)bh * S_ + s];
        const size_t orow = ((size_t)b_ * S_ + s) * D_ + (size_t)h * HD_;
        ctx[orow +  0 + lr] = (_Float16)(c0[r] * inv);
        ctx[orow + 16 + lr] = (_Float16)(c1[r] * inv);
        ctx[orow + 32 + lr] = (_Float16)(c2[r] * inv);
        ctx[orow + 48 + lr] = (_Float16)(c3[r] * inv);
    }
}

// ---------------- kernel 6: out_proj GEMM + bias -> f32 out ----------------
__global__ __launch_bounds__(256) void k_outproj(const _Float16* __restrict__ ctx,
                                                 const _Float16* __restrict__ Woh, // (D,D) f16
                                                 const float* __restrict__ bo,     // (D)
                                                 float* __restrict__ out) {
    const int NB = D_ / 64;                  // 8 col blocks
    const int nb = blockIdx.x % NB;
    const int mb = blockIdx.x / NB;
    const int row0 = mb * 128;
    const int col0 = nb * 64;

    v8f c[2][2] = {};
    gemm_block_128x64(ctx, D_, row0, Woh, D_, col0, D_, c);

    const int lane = threadIdx.x & 31;
    const int warp = threadIdx.x >> 5;
    const int hi = lane >> 4, lr = lane & 15;
    const int wm = warp >> 1, wn = warp & 1;

#pragma unroll
    for (int ni = 0; ni < 2; ++ni) {
        const int col = col0 + wn * 32 + ni * 16 + lr;
        const float bias = bo[col];
#pragma unroll
        for (int mi = 0; mi < 2; ++mi) {
#pragma unroll
            for (int r = 0; r < 8; ++r) {
                const int row = row0 + wm * 32 + mi * 16 + r + hi * 8;
                out[(size_t)row * D_ + col] = c[mi][ni][r] + bias;
            }
        }
    }
}

// ---------------- launch ----------------
extern "C" void kernel_launch(void* const* d_in, const int* in_sizes, int n_in,
                              void* d_out, int out_size, void* d_ws, size_t ws_size,
                              hipStream_t stream) {
    (void)in_sizes; (void)n_in; (void)out_size; (void)ws_size;
    const float* X          = (const float*)d_in[0];  // query (B,S,D)
    // d_in[1] key, d_in[2] value: unused by the reference module
    const unsigned char* mk = (const unsigned char*)d_in[3];
    const float* Wi         = (const float*)d_in[4];
    const float* bi         = (const float*)d_in[5];
    const float* Wo         = (const float*)d_in[6];
    const float* bo         = (const float*)d_in[7];
    float* out              = (float*)d_out;

    char* ws = (char*)d_ws;
    size_t off = 0;
    auto alloc = [&](size_t bytes) -> char* {
        char* p = ws + off;
        off += (bytes + 255) & ~(size_t)255;
        return p;
    };
    _Float16* Xh   = (_Float16*)alloc((size_t)BS_ * D_ * 2);            // 33.5 MB
    _Float16* Wih  = (_Float16*)alloc((size_t)TD_ * D_ * 2);            //  1.5 MB
    _Float16* Woh  = (_Float16*)alloc((size_t)D_ * D_ * 2);             //  0.5 MB
    _Float16* qh   = (_Float16*)alloc((size_t)B_ * H_ * S_ * HD_ * 2);  // 33.5 MB
    _Float16* kT   = (_Float16*)alloc((size_t)B_ * H_ * HD_ * S_ * 2);  // 33.5 MB
    _Float16* vT   = (_Float16*)alloc((size_t)B_ * H_ * HD_ * S_ * 2);  // 33.5 MB
    _Float16* ctx  = (_Float16*)alloc((size_t)BS_ * D_ * 2);            // 33.5 MB
    _Float16* kvT  = (_Float16*)alloc((size_t)B_ * H_ * HD_ * HD_ * 2);
    float*    ksum = (float*)   alloc((size_t)B_ * H_ * HD_ * 4);
    float*    den  = (float*)   alloc((size_t)B_ * H_ * S_ * 4);

    const int nX  = BS_ * D_;     // 16,777,216
    const int nWi = TD_ * D_;     //    786,432
    const int nWo = D_ * D_;      //    262,144
    k_cvt<<<nX  / 1024, 256, 0, stream>>>(X,  Xh,  nX);
    k_cvt<<<nWi / 1024, 256, 0, stream>>>(Wi, Wih, nWi);
    k_cvt<<<nWo / 1024, 256, 0, stream>>>(Wo, Woh, nWo);

    k_qkv    <<<(BS_ / 128) * (TD_ / 64), 256, 0, stream>>>(Xh, Wih, bi, mk, qh, kT, vT);
    k_ksum   <<<B_ * H_,                  256, 0, stream>>>(kT, ksum);
    k_kv     <<<B_ * H_,                  512, 0, stream>>>(kT, vT, kvT);
    k_denom  <<<(B_ * H_ * S_) / 256,     256, 0, stream>>>(qh, ksum, den);
    k_attn   <<<B_ * H_ * (S_ / 16) / 8,  256, 0, stream>>>(qh, kvT, den, ctx);
    k_outproj<<<(BS_ / 128) * (D_ / 64),  256, 0, stream>>>(ctx, Woh, bo, out);
}
